// TextFeat_MoEv1_89936615178774
// MI455X (gfx1250) — matrix operand order repaired and checked by reading, not compiled, not run on hardware
//
#include <hip/hip_runtime.h>
#include <hip/hip_bf16.h>
#include <math.h>

// Problem constants (from reference setup_inputs)
#define NB   2048      // N = B*T = 128*16
#define LT   64        // tokens per sequence
#define DV   512       // embedding dim
#define NE   8         // experts
#define NF   256       // output features
#define PADT 0
#define NEG_INF -1e30f
#define ATS  68        // f32 score row stride (bank-conflict-free, 16B aligned)
#define ATS2 72        // f16 attn row stride  (144B = 9*16B aligned)

typedef __attribute__((ext_vector_type(16))) _Float16 v16h;
typedef __attribute__((ext_vector_type(8)))  _Float16 v8h;
typedef __attribute__((ext_vector_type(4)))  _Float16 v4h;
typedef __attribute__((ext_vector_type(2)))  _Float16 v2h;
typedef __attribute__((ext_vector_type(8)))  float    v8f;

__device__ __forceinline__ v16h cat8(v8h x, v8h y) {
    return __builtin_shufflevector(x, y, 0,1,2,3,4,5,6,7,8,9,10,11,12,13,14,15);
}

// D += Ah*Bh + Ah*Bl + Al*Bh  (split-precision f16 product, ~22-bit mantissa)
__device__ __forceinline__ v8f wmma3(v16h ah, v16h al, v16h bh, v16h bl, v8f acc) {
    acc = __builtin_amdgcn_wmma_f32_16x16x32_f16(false, ah, false, bh, (short)0, acc, false, false);
    acc = __builtin_amdgcn_wmma_f32_16x16x32_f16(false, ah, false, bl, (short)0, acc, false, false);
    acc = __builtin_amdgcn_wmma_f32_16x16x32_f16(false, al, false, bh, (short)0, acc, false, false);
    return acc;
}

// ---------------------------------------------------------------------------
// Kernel 1a: proj[e][d] = sum_o Wk[e][o][d]*q[e][o], stored as hi/lo f16 planes
// ---------------------------------------------------------------------------
__global__ __launch_bounds__(256) void proj_fold_kernel(const float* __restrict__ Wk,
                                                        const float* __restrict__ q,
                                                        _Float16* __restrict__ proj_hi,
                                                        _Float16* __restrict__ proj_lo) {
    int idx = blockIdx.x * blockDim.x + threadIdx.x;   // 0..4095
    int e = idx >> 9;
    int d = idx & (DV - 1);
    const float* w  = Wk + (size_t)e * DV * DV + d;
    const float* qe = q + (size_t)e * DV;
    float s = 0.f;
    for (int o = 0; o < DV; ++o) s = fmaf(w[(size_t)o * DV], qe[o], s);
    _Float16 h = (_Float16)s;
    proj_hi[idx] = h;
    proj_lo[idx] = (_Float16)(s - (float)h);
}

// ---------------------------------------------------------------------------
// Kernel 1b: split trans_W into hi/lo f16 planes (once)
// ---------------------------------------------------------------------------
__global__ __launch_bounds__(256) void split_w_kernel(const float* __restrict__ W,
                                                      _Float16* __restrict__ w_hi,
                                                      _Float16* __restrict__ w_lo) {
    int idx = blockIdx.x * blockDim.x + threadIdx.x;   // NF*DV
    float x = W[idx];
    _Float16 h = (_Float16)x;
    w_hi[idx] = h;
    w_lo[idx] = (_Float16)(x - (float)h);
}

// ---------------------------------------------------------------------------
// Kernel 2: per-sequence MoE attention pooling. Token tile lives in LDS as
// pre-split hi/lo f16 planes (converted once during the gather); all WMMA
// K-loops are pure b128 loads + v_wmma. Gate mean is fused into the gather
// (each thread's iterations touch a fixed d-range since 256 = 0 mod 128).
// Fragment rows/cols e>=8 carry duplicated expert data (outputs discarded).
// ---------------------------------------------------------------------------
__global__ __launch_bounds__(256) void moe_pool_kernel(const int*   __restrict__ sample,
                                                       const float* __restrict__ emb,
                                                       const _Float16* __restrict__ proj_hi,
                                                       const _Float16* __restrict__ proj_lo,
                                                       const float* __restrict__ gate_W,
                                                       const float* __restrict__ gate_b,
                                                       _Float16* __restrict__ pool_hi,
                                                       _Float16* __restrict__ pool_lo) {
    extern __shared__ char smem_raw[];
    _Float16* tok_hi = (_Float16*)smem_raw;            // 64*512 halves (64KB)
    _Float16* tok_lo = tok_hi + LT * DV;               // 64*512 halves (64KB)
    _Float16* at_hi  = tok_lo + LT * DV;               // 8*ATS2
    _Float16* at_lo  = at_hi + NE * ATS2;              // 8*ATS2
    float*    scT    = (float*)(at_lo + NE * ATS2);    // 8*ATS f32 scores^T
    float*    eout   = scT + NE * ATS;                 // 8*512
    float*    gpart  = eout + NE * DV;                 // 256*4 gate partials
    float*    gate_in= gpart + 1024;                   // 512
    float*    red    = gate_in + DV;                   // 16

    __shared__ int toks[LT];
    __shared__ int s_len;

    const int n    = blockIdx.x;
    const int t    = threadIdx.x;
    const int wave = t >> 5;
    const int lane = t & 31;
    const int grp  = lane >> 4;       // 0 | 1 (half-wave)
    const int l15  = lane & 15;

    if (t < LT) toks[t] = sample[(size_t)n * LT + t];
    __syncthreads();

    // ---- gather: global f32 -> LDS (hi,lo) f16 planes + gate partials ----
    {
        const int NV4 = LT * DV / 4;                  // 8192 vec4
        float gx = 0.f, gy = 0.f, gz = 0.f, gw = 0.f;
        for (int v = t; v < NV4; v += 256) {
            int l   = v >> 7;                         // 128 vec4 per token row
            int tok = toks[l];
            int vn   = (v + 256 < NV4) ? v + 256 : v;
            int tokn = toks[vn >> 7];
            __builtin_prefetch(emb + (size_t)tokn * DV + (size_t)(vn & 127) * 4, 0, 3);
            float4 val = {0.f, 0.f, 0.f, 0.f};
            if (tok != PADT)                          // wave-uniform (row per wave)
                val = *(const float4*)(emb + (size_t)tok * DV + (size_t)(v & 127) * 4);
            gx += val.x; gy += val.y; gz += val.z; gw += val.w;
            _Float16 h0 = (_Float16)val.x, h1 = (_Float16)val.y,
                     h2 = (_Float16)val.z, h3 = (_Float16)val.w;
            v4h hv = {h0, h1, h2, h3};
            v4h lv = {(_Float16)(val.x - (float)h0), (_Float16)(val.y - (float)h1),
                      (_Float16)(val.z - (float)h2), (_Float16)(val.w - (float)h3)};
            *(v4h*)(tok_hi + (size_t)v * 4) = hv;
            *(v4h*)(tok_lo + (size_t)v * 4) = lv;
        }
        float4 gp = {gx, gy, gz, gw};
        *(float4*)(gpart + (size_t)t * 4) = gp;
    }
    if (t == 0) {
        int c = 0;
        for (int l = 0; l < LT; ++l) c += (toks[l] != PADT);
        s_len = c;
    }
    __syncthreads();
    const bool all_pad = (s_len == 0);

    // ---- Phase B: scores = tok(64x512) @ proj^T(512x16) via WMMA (waves 0-3)
    //      waves 4-7: finish gate_in reduction (2-way over row-parity partials)
    if (wave < 4) {
        const int arow = wave * 16 + l15;                       // token row
        const _Float16* Ah = tok_hi + (size_t)arow * DV;
        const _Float16* Al = tok_lo + (size_t)arow * DV;
        const _Float16* Bh = proj_hi + (size_t)(l15 & 7) * DV;  // dup for l15>=8
        const _Float16* Bl = proj_lo + (size_t)(l15 & 7) * DV;
        v8f acc = {};
        for (int k = 0; k < DV; k += 32) {
            v16h ah = cat8(*(const v8h*)(Ah + k + 8 * grp), *(const v8h*)(Ah + k + 16 + 8 * grp));
            v16h al = cat8(*(const v8h*)(Al + k + 8 * grp), *(const v8h*)(Al + k + 16 + 8 * grp));
            v16h bh = cat8(*(const v8h*)(Bh + k + 8 * grp), *(const v8h*)(Bh + k + 16 + 8 * grp));
            v16h bl = cat8(*(const v8h*)(Bl + k + 8 * grp), *(const v8h*)(Bl + k + 16 + 8 * grp));
            acc = wmma3(ah, al, bh, bl, acc);
        }
        if (l15 < 8) {                       // store masked scores^T (f32)
#pragma unroll
            for (int j = 0; j < 8; ++j) {
                int l = wave * 16 + j + 8 * grp;
                int tokl = toks[l];
                float s = acc[j];
                if (all_pad)           s = 1.0f;
                else if (tokl == PADT) s = NEG_INF;
                scT[l15 * ATS + l] = s;
            }
        }
    } else if (t >= 128) {
        const int i = t - 128;               // d-block 0..127
        float4 a = *(const float4*)(gpart + (size_t)i * 4);        // even rows
        float4 b = *(const float4*)(gpart + (size_t)t * 4);        // odd rows
        float4 g = {(a.x + b.x) * (1.f / LT), (a.y + b.y) * (1.f / LT),
                    (a.z + b.z) * (1.f / LT), (a.w + b.w) * (1.f / LT)};
        *(float4*)(gate_in + (size_t)i * 4) = g;
    }
    __syncthreads();

    // ---- Phase C: softmax per expert -> split attn planes (t<8)
    //              || gate scores (wave 1 lanes 0-7) ----
    if (t < NE) {
        float* r = scT + t * ATS;
        float m = -3.4e38f;
        for (int l = 0; l < LT; ++l) m = fmaxf(m, r[l]);
        float sum = 0.f;
        for (int l = 0; l < LT; ++l) { float ex = __expf(r[l] - m); r[l] = ex; sum += ex; }
        float inv = 1.f / sum;
        for (int l = 0; l < LT; ++l) {
            float wv = r[l] * inv;
            _Float16 h = (_Float16)wv;
            at_hi[t * ATS2 + l] = h;
            at_lo[t * ATS2 + l] = (_Float16)(wv - (float)h);
        }
    } else if (t >= 32 && t < 32 + NE) {
        int e = t - 32;
        float s = gate_b[e];
        const float* gw = gate_W + (size_t)e * DV;
        for (int d = 0; d < DV; ++d) s = fmaf(gate_in[d], gw[d], s);
        red[e] = s;
    }
    __syncthreads();

    // ---- Phase D: expert_out = attn^T(16x64) @ tok(64x512) via WMMA ----
    {
        const _Float16* arh = at_hi + (size_t)(l15 & 7) * ATS2;  // rows 8..15 dup
        const _Float16* arl = at_lo + (size_t)(l15 & 7) * ATS2;
        v16h a0h = cat8(*(const v8h*)(arh + 16 * grp),      *(const v8h*)(arh + 8 + 16 * grp));
        v16h a0l = cat8(*(const v8h*)(arl + 16 * grp),      *(const v8h*)(arl + 8 + 16 * grp));
        v16h a1h = cat8(*(const v8h*)(arh + 32 + 16 * grp), *(const v8h*)(arh + 40 + 16 * grp));
        v16h a1l = cat8(*(const v8h*)(arl + 32 + 16 * grp), *(const v8h*)(arl + 40 + 16 * grp));
        for (int dt = wave; dt < DV / 16; dt += 8) {
            const int col = dt * 16 + l15;
            v8f acc = {};
            v16h bh, bl;
#pragma unroll
            for (int j = 0; j < 16; ++j) {
                bh[j] = tok_hi[(16 * grp + j) * DV + col];
                bl[j] = tok_lo[(16 * grp + j) * DV + col];
            }
            acc = wmma3(a0h, a0l, bh, bl, acc);
#pragma unroll
            for (int j = 0; j < 16; ++j) {
                bh[j] = tok_hi[(32 + 16 * grp + j) * DV + col];
                bl[j] = tok_lo[(32 + 16 * grp + j) * DV + col];
            }
            acc = wmma3(a1h, a1l, bh, bl, acc);
            if (grp == 0) {                  // rows 0..7 = valid experts
#pragma unroll
                for (int j = 0; j < 8; ++j) eout[j * DV + col] = acc[j];
            }
        }
    }
    __syncthreads();

    // ---- Phase E: top-2 gate (redundant, uniform) + pooled split store ----
    {
        float v0 = red[0]; int i0 = 0;
        for (int e = 1; e < NE; ++e) { float v = red[e]; if (v > v0) { v0 = v; i0 = e; } }
        float v1 = -3.4e38f; int i1 = 0;
        for (int e = 0; e < NE; ++e) { float v = red[e]; if (e != i0 && v > v1) { v1 = v; i1 = e; } }
        float e1  = __expf(v1 - v0);
        float inv = 1.f / (1.f + e1);
        float w0 = inv, w1 = e1 * inv;
        const int d0 = t * 2;                // DV/2 == 256 == blockDim
        float px = w0 * eout[i0 * DV + d0]     + w1 * eout[i1 * DV + d0];
        float py = w0 * eout[i0 * DV + d0 + 1] + w1 * eout[i1 * DV + d0 + 1];
        if (all_pad) { px = 0.f; py = 0.f; }
        _Float16 hx = (_Float16)px, hy = (_Float16)py;
        v2h hv = {hx, hy};
        v2h lv = {(_Float16)(px - (float)hx), (_Float16)(py - (float)hy)};
        *(v2h*)(pool_hi + (size_t)n * DV + d0) = hv;
        *(v2h*)(pool_lo + (size_t)n * DV + d0) = lv;
    }
}

// ---------------------------------------------------------------------------
// Kernel 3: out = gelu(pooled @ trans_W^T + b), all operands pre-split.
// One wave per 16x16 tile; K=512 in 16 steps: 8 b128 loads + 3 v_wmma each.
// ---------------------------------------------------------------------------
__global__ __launch_bounds__(32) void transform_wmma_kernel(const _Float16* __restrict__ pool_hi,
                                                            const _Float16* __restrict__ pool_lo,
                                                            const _Float16* __restrict__ w_hi,
                                                            const _Float16* __restrict__ w_lo,
                                                            const float* __restrict__ bias,
                                                            float* __restrict__ out) {
    const int m0   = blockIdx.x << 4;
    const int f0   = blockIdx.y << 4;
    const int lane = threadIdx.x;
    const int grp  = lane >> 4;
    const int l15  = lane & 15;

    const _Float16* Ah = pool_hi + (size_t)(m0 + l15) * DV;
    const _Float16* Al = pool_lo + (size_t)(m0 + l15) * DV;
    const _Float16* Bh = w_hi + (size_t)(f0 + l15) * DV;    // B[k,f] = W[f,k]
    const _Float16* Bl = w_lo + (size_t)(f0 + l15) * DV;

    v8f acc = {};
    for (int k = 0; k < DV; k += 32) {
        v16h ah = cat8(*(const v8h*)(Ah + k + 8 * grp), *(const v8h*)(Ah + k + 16 + 8 * grp));
        v16h al = cat8(*(const v8h*)(Al + k + 8 * grp), *(const v8h*)(Al + k + 16 + 8 * grp));
        v16h bh = cat8(*(const v8h*)(Bh + k + 8 * grp), *(const v8h*)(Bh + k + 16 + 8 * grp));
        v16h bl = cat8(*(const v8h*)(Bl + k + 8 * grp), *(const v8h*)(Bl + k + 16 + 8 * grp));
        acc = wmma3(ah, al, bh, bl, acc);
    }

    const int   col = f0 + l15;
    const float bc  = bias[col];
#pragma unroll
    for (int j = 0; j < 8; ++j) {
        int row = m0 + j + 8 * grp;
        float x = acc[j] + bc;
        float g = 0.5f * x * (1.0f + erff(x * 0.70710678118654752f));
        out[(size_t)row * NF + col] = g;
    }
}

// ---------------------------------------------------------------------------
extern "C" void kernel_launch(void* const* d_in, const int* in_sizes, int n_in,
                              void* d_out, int out_size, void* d_ws, size_t ws_size,
                              hipStream_t stream) {
    (void)in_sizes; (void)n_in; (void)out_size; (void)ws_size;
    const int*   sample = (const int*)  d_in[0];
    const float* emb    = (const float*)d_in[1];
    const float* eq     = (const float*)d_in[2];
    const float* eWk    = (const float*)d_in[3];
    const float* gW     = (const float*)d_in[4];
    const float* gb     = (const float*)d_in[5];
    const float* tW     = (const float*)d_in[6];
    const float* tb     = (const float*)d_in[7];
    float*       out    = (float*)d_out;

    _Float16* proj_hi = (_Float16*)d_ws;           // 8*512
    _Float16* proj_lo = proj_hi + NE * DV;         // 8*512
    _Float16* tw_hi   = proj_lo + NE * DV;         // 256*512
    _Float16* tw_lo   = tw_hi + (size_t)NF * DV;   // 256*512
    _Float16* pool_hi = tw_lo + (size_t)NF * DV;   // 2048*512
    _Float16* pool_lo = pool_hi + (size_t)NB * DV; // 2048*512  (total ~4.6 MB)

    proj_fold_kernel<<<(NE * DV) / 256, 256, 0, stream>>>(eWk, eq, proj_hi, proj_lo);
    split_w_kernel<<<(NF * DV) / 256, 256, 0, stream>>>(tW, tw_hi, tw_lo);

    size_t smem = (size_t)(2 * LT * DV + 2 * NE * ATS2) * sizeof(_Float16)
                + (size_t)(NE * ATS + NE * DV + 1024 + DV + 16) * sizeof(float);
    moe_pool_kernel<<<NB, 256, smem, stream>>>(sample, emb, proj_hi, proj_lo,
                                               gW, gb, pool_hi, pool_lo);

    transform_wmma_kernel<<<dim3(NB / 16, NF / 16), 32, 0, stream>>>(pool_hi, pool_lo,
                                                                     tw_hi, tw_lo, tb, out);
}
